// Encoder_85581518340599
// MI455X (gfx1250) — matrix operand order repaired and checked by reading, not compiled, not run on hardware
//
#include <hip/hip_runtime.h>

// ---------------------------------------------------------------------------
// Masked LSTM encoder scan for MI455X (gfx1250, wave32, WMMA).
//   B=512, T=256, F=128, L=256, 4L=1024.
// Each workgroup (512 threads = 16 waves) owns 16 batch rows and runs the
// whole T=256 scan locally (rows are independent through the recurrence).
// x@W fused per-step; bf16 WMMA 16x16x32 with f32 accumulation; c/h state in
// VGPRs. W/U pre-swizzled to a B-fragment layout with k = 16*half + i -> each fragment is
// one contiguous 32B/lane load, streamed from L2 every step.
// The `opaque` asm token stops the compiler from hoisting the loop-invariant
// weight loads out of the t-loop (which previously spilled 3KB/lane to
// scratch and turned the L2 stream into scratch reloads).
// ---------------------------------------------------------------------------

typedef float  v8f   __attribute__((ext_vector_type(8)));
typedef __bf16 v16bf __attribute__((ext_vector_type(16)));
typedef unsigned short v16u __attribute__((ext_vector_type(16)));

#define B_DIM 512
#define T_DIM 256
#define F_DIM 128
#define L_DIM 256
#define W_ELEMS (F_DIM * 4 * L_DIM)   // 131072
#define U_ELEMS (L_DIM * 4 * L_DIM)   // 262144

__device__ __forceinline__ unsigned short f2bf(float f) {
  unsigned int u = __float_as_uint(f);
  u += 0x7FFFu + ((u >> 16) & 1u);          // round-to-nearest-even
  return (unsigned short)(u >> 16);
}
__device__ __forceinline__ unsigned int f2bf2(float lo, float hi) {
  return (unsigned int)f2bf(lo) | ((unsigned int)f2bf(hi) << 16);
}
__device__ __forceinline__ float sigmoidf(float x) {
  return 1.0f / (1.0f + __expf(-x));
}
// A-fragment K-within-block mapping: a[i] holds K = i + 16*half.
// Inverse (given kk = K%32): half = kk>>4 ; i = kk & 15
__device__ __forceinline__ void a_inv(int kk, int& hh, int& ii) {
  hh = kk >> 4;
  ii = kk & 15;
}

// ---------------------------------------------------------------------------
// Prep: swizzle W [128,1024] and U [256,1024] (f32) into bf16 B-fragment
// layout: frag[(nt*KB + kb)*32 + lane][i] = M[32*kb + i + 16*(lane/16)][16*nt + lane%16]
// ---------------------------------------------------------------------------
__global__ __launch_bounds__(256) void prep_weights_kernel(
    const float* __restrict__ W, const float* __restrict__ U,
    unsigned short* __restrict__ Wsw, unsigned short* __restrict__ Usw)
{
  int idx = blockIdx.x * 256 + threadIdx.x;
  if (idx < W_ELEMS) {
    int i = idx & 15, lane = (idx >> 4) & 31, kb = (idx >> 9) & 3, nt = idx >> 11;
    int K = (kb << 5) + i + ((lane >> 4) << 4);
    int N = (nt << 4) + (lane & 15);
    Wsw[idx] = f2bf(W[K * 1024 + N]);
  } else if (idx < W_ELEMS + U_ELEMS) {
    int j = idx - W_ELEMS;
    int i = j & 15, lane = (j >> 4) & 31, kb = (j >> 9) & 7, nt = j >> 12;
    int K = (kb << 5) + i + ((lane >> 4) << 4);
    int N = (nt << 4) + (lane & 15);
    Usw[j] = f2bf(U[K * 1024 + N]);
  }
}

// ---------------------------------------------------------------------------
// Main persistent scan kernel. grid = 32 (B/16), block = 512 (16 waves).
// ---------------------------------------------------------------------------
__global__ __launch_bounds__(512, 1) void lstm_scan_kernel(
    const float* __restrict__ x, const unsigned short* __restrict__ Wsw,
    const unsigned short* __restrict__ Usw, const float* __restrict__ bias,
    float* __restrict__ out)
{
  __shared__ __attribute__((aligned(32))) unsigned short x_swz[4][32][16]; // 4KB  A-frags of x_t
  __shared__ __attribute__((aligned(32))) unsigned short h_swz[8][32][16]; // 8KB  A-frags of h
  __shared__ int   mask_lds[2][16];                                        // double-buffered row mask
  __shared__ __attribute__((aligned(16))) float h_out[16][256];            // epilogue staging

  const int tid  = threadIdx.x;
  const int w    = tid >> 5;        // wave 0..15: owns columns l in [16w,16w+16)
  const int lane = tid & 31;
  const int half = lane >> 4;
  const int l16  = lane & 15;
  const int b0   = blockIdx.x << 4;

  // zero h_swz (h0 = 0) and both mask buffers
  unsigned int* hz = (unsigned int*)&h_swz[0][0][0];
  for (int i = tid; i < 2048; i += 512) hz[i] = 0u;
  if (tid < 16) { mask_lds[0][tid] = 0; mask_lds[1][tid] = 0; }

  // per-lane bias for each gate: column = g*256 + 16w + l16
  float bia[4];
  #pragma unroll
  for (int g = 0; g < 4; ++g) bia[g] = bias[g * 256 + (w << 4) + l16];

  v8f c_reg = {0.f,0.f,0.f,0.f,0.f,0.f,0.f,0.f};
  v8f h_reg = {0.f,0.f,0.f,0.f,0.f,0.f,0.f,0.f};

  // constant swizzle coords for writing h (K = l = 16w + l16)
  const int h_kb = w >> 1;
  const int h_kk = ((w & 1) << 4) | l16;
  int h_hh, h_ii; a_inv(h_kk, h_hh, h_ii);

  // x staging swizzle coords (wave w stages batch row w; f = 4*lane)
  const int x_f  = 4 * lane;
  const int x_kb = x_f >> 5;
  int x_hh, x_ii; a_inv(x_f & 31, x_hh, x_ii);
  const int x_lt = w + 16 * x_hh;

  __syncthreads();

  // stage x(t=0) + mask(t=0)
  {
    float4 xa = *(const float4*)(x + ((size_t)(b0 + w) * T_DIM + 0) * F_DIM + x_f);
    *(unsigned int*)&x_swz[x_kb][x_lt][x_ii]     = f2bf2(xa.x, xa.y);
    *(unsigned int*)&x_swz[x_kb][x_lt][x_ii + 2] = f2bf2(xa.z, xa.w);
    int nz = (xa.x != 0.f || xa.y != 0.f || xa.z != 0.f || xa.w != 0.f) ? 1 : 0;
    atomicOr(&mask_lds[0][w], nz);
  }

  // Opaque zero: redefined by volatile asm every iteration so the compiler
  // cannot hoist/CSE the (address-invariant) weight loads across timesteps.
  unsigned int opaque = 0u;

  for (int t = 0; t < T_DIM; ++t) {
    const int p = t & 1;
    __syncthreads();                         // barrier A: h_swz/x_swz/mask(t) visible
    if (tid < 16) mask_lds[1 - p][tid] = 0;  // reset next-step mask buffer

    asm volatile("" : "+s"(opaque));         // fresh value each iteration (really always 0)
    const unsigned short* Wp = Wsw + opaque;
    const unsigned short* Up = Usw + opaque;

    // prefetch next timestep's input early (hidden under the GEMM)
    float4 xn = {0.f, 0.f, 0.f, 0.f};
    if (t + 1 < T_DIM)
      xn = *(const float4*)(x + ((size_t)(b0 + w) * T_DIM + (t + 1)) * F_DIM + x_f);

    // z = bias
    v8f acc[4];
    #pragma unroll
    for (int g = 0; g < 4; ++g) {
      #pragma unroll
      for (int r = 0; r < 8; ++r) acc[g][r] = bia[g];
    }

    // z += x_t @ W   (K = 128 -> 4 k-blocks)
    #pragma unroll
    for (int kb = 0; kb < 4; ++kb) {
      v16bf a = __builtin_bit_cast(v16bf, *(const v16u*)&x_swz[kb][lane][0]);
      v16u bu[4];
      #pragma unroll
      for (int g = 0; g < 4; ++g)
        bu[g] = *(const v16u*)(Wp + (size_t)(((((g << 4) + w) * 4 + kb) * 32 + lane) * 16));
      #pragma unroll
      for (int g = 0; g < 4; ++g)
        acc[g] = __builtin_amdgcn_wmma_f32_16x16x32_bf16(
            false, a, false, __builtin_bit_cast(v16bf, bu[g]), (short)0, acc[g], false, false);
    }
    // z += h_{t-1} @ U   (K = 256 -> 8 k-blocks)
    #pragma unroll
    for (int kb = 0; kb < 8; ++kb) {
      v16bf a = __builtin_bit_cast(v16bf, *(const v16u*)&h_swz[kb][lane][0]);
      v16u bu[4];
      #pragma unroll
      for (int g = 0; g < 4; ++g)
        bu[g] = *(const v16u*)(Up + (size_t)(((((g << 4) + w) * 8 + kb) * 32 + lane) * 16));
      #pragma unroll
      for (int g = 0; g < 4; ++g)
        acc[g] = __builtin_amdgcn_wmma_f32_16x16x32_bf16(
            false, a, false, __builtin_bit_cast(v16bf, bu[g]), (short)0, acc[g], false, false);
    }

    // gates (all four gates for column l live in this wave's registers)
    #pragma unroll
    for (int r = 0; r < 8; ++r) {
      int row = r + 8 * half;                // C layout: M = r + 8*(lane/16)
      float ig = sigmoidf(acc[0][r]);
      float fg = sigmoidf(acc[1][r]);
      float gg = fmaxf(acc[2][r], 0.0f);     // relu candidate
      float og = sigmoidf(acc[3][r]);
      float cn = fg * c_reg[r] + ig * gg;
      float hn = og * fmaxf(cn, 0.0f);       // relu on cell output
      bool  m  = mask_lds[p][row] != 0;      // masked steps carry state
      c_reg[r] = m ? cn : c_reg[r];
      h_reg[r] = m ? hn : h_reg[r];
    }

    __syncthreads();                         // barrier B: GEMM reads of h_swz done

    // publish h_t as bf16 A-fragments for next step's h@U
    #pragma unroll
    for (int r = 0; r < 8; ++r) {
      int row = r + 8 * half;
      h_swz[h_kb][row + 16 * h_hh][h_ii] = f2bf(h_reg[r]);
    }
    // stage x(t+1) + its mask
    if (t + 1 < T_DIM) {
      *(unsigned int*)&x_swz[x_kb][x_lt][x_ii]     = f2bf2(xn.x, xn.y);
      *(unsigned int*)&x_swz[x_kb][x_lt][x_ii + 2] = f2bf2(xn.z, xn.w);
      int nz = (xn.x != 0.f || xn.y != 0.f || xn.z != 0.f || xn.w != 0.f) ? 1 : 0;
      atomicOr(&mask_lds[1 - p][w], nz);
    }
  }

  // Epilogue: RepeatVector — out[b, t, :] = h_last[b, :] for all t.
  __syncthreads();
  #pragma unroll
  for (int r = 0; r < 8; ++r)
    h_out[r + 8 * half][(w << 4) + l16] = h_reg[r];
  __syncthreads();

  for (int it = tid; it < 16 * 256 * 64; it += 512) {
    int l4  = it & 63;
    int tt  = (it >> 6) & 255;
    int row = it >> 14;
    float4 v = *(const float4*)&h_out[row][l4 << 2];
    *(float4*)(out + (((size_t)(b0 + row) * T_DIM + tt) << 8) + (l4 << 2)) = v;
  }
}

// ---------------------------------------------------------------------------
extern "C" void kernel_launch(void* const* d_in, const int* in_sizes, int n_in,
                              void* d_out, int out_size, void* d_ws, size_t ws_size,
                              hipStream_t stream) {
  const float* x  = (const float*)d_in[0];   // [512, 256, 128]
  const float* W  = (const float*)d_in[1];   // [128, 1024]
  const float* U  = (const float*)d_in[2];   // [256, 1024]
  const float* b  = (const float*)d_in[3];   // [1024]
  float*       out = (float*)d_out;          // [512, 256, 256]

  unsigned short* Wsw = (unsigned short*)d_ws;          // 256 KB
  unsigned short* Usw = Wsw + W_ELEMS;                  // 512 KB

  prep_weights_kernel<<<(W_ELEMS + U_ELEMS) / 256, 256, 0, stream>>>(W, U, Wsw, Usw);
  lstm_scan_kernel<<<B_DIM / 16, 512, 0, stream>>>(x, Wsw, Usw, b, out);
}
